// HAN_68702296867438
// MI455X (gfx1250) — matrix operand order repaired and checked by reading, not compiled, not run on hardware
//
#include <hip/hip_runtime.h>
#include <hip/hip_bf16.h>

// ---------------------------------------------------------------------------
// Types for CDNA5 WMMA
// ---------------------------------------------------------------------------
typedef __attribute__((ext_vector_type(16))) __bf16 v16bf;
typedef __attribute__((ext_vector_type(8)))  float  v8f;

union BF16Frag { v16bf v; uint4 q[2]; unsigned short s[16]; };

__device__ __forceinline__ unsigned short f2bf(float f) {
    // round-to-nearest-even fp32 -> bf16
    unsigned u = __float_as_uint(f);
    u += 0x7FFFu + ((u >> 16) & 1u);
    return (unsigned short)(u >> 16);
}

// order-preserving float -> uint key (for atomic max on signed floats)
__device__ __forceinline__ unsigned fkey(float f) {
    unsigned u = __float_as_uint(f);
    return (u & 0x80000000u) ? ~u : (u | 0x80000000u);
}
__device__ __forceinline__ float funkey(unsigned k) {
    return (k & 0x80000000u) ? __uint_as_float(k & 0x7FFFFFFFu) : __uint_as_float(~k);
}

// branch-free tanh via exp: tanh(x) = 1 - 2/(exp(2x)+1); exact limits at +/-inf
__device__ __forceinline__ float fast_tanh(float x) {
    return 1.f - 2.f / (__expf(2.f * x) + 1.f);
}

// ---------------------------------------------------------------------------
// One-time weight pre-pack: f32 [K,256] row-major -> bf16 in WMMA B-fragment
// order.  Flat layout: [tile = ktile*16 + coltile][lane 0..31][elem 0..15],
// where for lane: n = coltile*16 + (lane&15), half = lane>>4, and
// elem e -> v = e>>1, pair = e&1, k = ktile*32 + 16*half + 2*v + pair.
// A fragment load then becomes two b128 loads with zero conversion VALU.
// ---------------------------------------------------------------------------
__global__ void pack_w_bf16(const float* __restrict__ W, unsigned short* __restrict__ Wp, int n)
{
    int idx = blockIdx.x * blockDim.x + threadIdx.x;
    if (idx >= n) return;
    const int e       = idx & 15;
    const int lane    = (idx >> 4) & 31;
    const int tile    = idx >> 9;
    const int ktile   = tile >> 4;
    const int coltile = tile & 15;
    const int half    = lane >> 4;
    const int ncol    = (coltile << 4) + (lane & 15);
    const int k       = (ktile << 5) + (half << 4) + ((e >> 1) << 1) + (e & 1);
    Wp[idx] = f2bf(W[(size_t)k * 256 + ncol]);
}

// ---------------------------------------------------------------------------
// WMMA GEMM:  C[M,256] = act(A[M,K]) @ W[K,256] (+bias)
//   mode 0: store C = acc + bias
//   mode 1: tanh-reduce epilogue: wred[col] += scale * sum_rows tanh(acc+bias)
//   reluA : apply relu to A elements while staging
// Block = 128 threads (4 waves); wave w owns a 16x64 tile (4 accumulators).
// A tile (16 x K) is converted to bf16 ONCE into LDS in A-fragment order
// (shared by all 4 waves); B comes pre-packed in fragment order from global.
// Inner loop: 2x ds_load_b128 + 8x global_load_b128 + 4x v_wmma per K-step.
// ---------------------------------------------------------------------------
__global__ __launch_bounds__(128)
void wmma_gemm_256(const float* __restrict__ A, const unsigned short* __restrict__ Wp,
                   const float* __restrict__ bias, float* __restrict__ C,
                   float* __restrict__ wred, int M, int K,
                   int reluA, int mode, float scale)
{
    __shared__ uint4 sA[8 * 32 * 2];   // up to K=256: 8 ktiles * 32 lanes * 32B

    const int lane = threadIdx.x & 31;
    const int wave = threadIdx.x >> 5;
    const int row0 = blockIdx.x << 4;
    if (row0 >= M) return;
    const int col0 = wave << 6;        // 64 cols per wave
    const int mrow = lane & 15;
    const int half = lane >> 4;
    const int ktiles = K >> 5;

    // ---- stage A tile into LDS as bf16, already in A-fragment lane order:
    // elem e of lane L in ktile kt holds A[row0 + (L&15)][k], with
    // k = kt*32 + 16*(L>>4) + e.
    {
        unsigned short* sA16 = (unsigned short*)sA;
        const int total = ktiles << 9;             // ktiles*32*16
        for (int t = threadIdx.x; t < total; t += 128) {
            const int e  = t & 15;
            const int ln = (t >> 4) & 31;
            const int kt = t >> 9;
            const int v  = e >> 1;
            const int k  = (kt << 5) + ((ln >> 4) << 4) + e;
            int r = row0 + (ln & 15); if (r >= M) r = M - 1;
            float a = A[(size_t)r * K + k];
            if (reluA) a = fmaxf(a, 0.f);
            sA16[t] = f2bf(a);
        }
    }
    __syncthreads();

    v8f acc[4];
    #pragma unroll
    for (int c = 0; c < 4; ++c) acc[c] = (v8f){0.f,0.f,0.f,0.f,0.f,0.f,0.f,0.f};

    for (int kt = 0; kt < ktiles; ++kt) {
        BF16Frag af;
        const uint4* ap = sA + (((kt << 5) + lane) << 1);
        af.q[0] = ap[0];
        af.q[1] = ap[1];

        #pragma unroll
        for (int c = 0; c < 4; ++c) {
            const int coltile = (col0 >> 4) + c;
            BF16Frag bf;
            const uint4* bp = (const uint4*)Wp
                            + ((((size_t)((kt << 4) + coltile)) << 5) + lane) * 2;
            bf.q[0] = bp[0];
            bf.q[1] = bp[1];
            acc[c] = __builtin_amdgcn_wmma_f32_16x16x32_bf16(
                         false, af.v, false, bf.v, (short)0, acc[c], false, false);
        }
    }

    if (mode == 0) {
        #pragma unroll
        for (int c = 0; c < 4; ++c) {
            const int ncol = col0 + (c << 4) + mrow;
            const float b = bias ? bias[ncol] : 0.f;
            #pragma unroll
            for (int i = 0; i < 8; ++i) {
                const int r = row0 + i + (half << 3);   // D: VGPR i -> row i + 8*half
                if (r < M) C[(size_t)r * 256 + ncol] = acc[c][i] + b;
            }
        }
    } else {
        // fused semantic-attention reduction: mean over rows of tanh(x @ kW + kB)
        #pragma unroll
        for (int c = 0; c < 4; ++c) {
            const int ncol = col0 + (c << 4) + mrow;
            const float b = bias[ncol];
            float s = 0.f;
            #pragma unroll
            for (int i = 0; i < 8; ++i) s += fast_tanh(acc[c][i] + b);
            s += __shfl_xor(s, 16, 32);                 // merge both 8-row halves
            if (half == 0) atomicAdd(wred + ncol, s * scale);
        }
    }
}

// ---------------------------------------------------------------------------
// Per-node attention logits: a[n][h] = sum_d xp[n][h*128+d] * lin[h][d]
// One wave per node; lanes 0-15 cover head 0, lanes 16-31 head 1.
// ---------------------------------------------------------------------------
__global__ __launch_bounds__(256)
void node_alpha(const float* __restrict__ xp, const float* __restrict__ lin,
                float* __restrict__ a, int N)
{
    const int node = blockIdx.x * (blockDim.x >> 5) + (threadIdx.x >> 5);
    const int lane = threadIdx.x & 31;
    if (node >= N) return;
    const float4* row = (const float4*)(xp + (size_t)node * 256);
    const float4* lw  = (const float4*)lin;
    float4 x0 = row[2*lane], x1 = row[2*lane+1];
    float4 w0 = lw [2*lane], w1 = lw [2*lane+1];
    float s = x0.x*w0.x + x0.y*w0.y + x0.z*w0.z + x0.w*w0.w
            + x1.x*w1.x + x1.y*w1.y + x1.z*w1.z + x1.w*w1.w;
    #pragma unroll
    for (int off = 1; off < 16; off <<= 1) s += __shfl_xor(s, off, 32);
    if ((lane & 15) == 0) a[(size_t)node * 2 + (lane >> 4)] = s;
}

__global__ void fill_f32(float* __restrict__ p, float v, int n) {
    int i = blockIdx.x * blockDim.x + threadIdx.x;
    if (i < n) p[i] = v;
}
__global__ void fill_u32(unsigned* __restrict__ p, unsigned v, int n) {
    int i = blockIdx.x * blockDim.x + threadIdx.x;
    if (i < n) p[i] = v;
}

// segment max of leaky_relu(a_src[si] + a_dst[di]) per (dst, head)
__global__ void edge_max_k(const int* __restrict__ si, const int* __restrict__ di,
                           const float* __restrict__ as, const float* __restrict__ ad,
                           unsigned* __restrict__ mx, int E)
{
    int e = blockIdx.x * blockDim.x + threadIdx.x;
    if (e >= E) return;
    const int s = si[e], d = di[e];
    #pragma unroll
    for (int h = 0; h < 2; ++h) {
        float v = as[(size_t)s*2 + h] + ad[(size_t)d*2 + h];
        v = v > 0.f ? v : 0.2f * v;
        atomicMax(mx + (size_t)d*2 + h, fkey(v));
    }
}

// alpha_exp = exp(v - segmax); segsum += alpha_exp
__global__ void edge_expsum_k(const int* __restrict__ si, const int* __restrict__ di,
                              const float* __restrict__ as, const float* __restrict__ ad,
                              const unsigned* __restrict__ mx, float* __restrict__ sum,
                              float* __restrict__ alpha, int E)
{
    int e = blockIdx.x * blockDim.x + threadIdx.x;
    if (e >= E) return;
    const int s = si[e], d = di[e];
    #pragma unroll
    for (int h = 0; h < 2; ++h) {
        float v = as[(size_t)s*2 + h] + ad[(size_t)d*2 + h];
        v = v > 0.f ? v : 0.2f * v;
        const float m  = funkey(mx[(size_t)d*2 + h]);
        const float ex = __expf(v - m);
        alpha[(size_t)e*2 + h] = ex;
        atomicAdd(sum + (size_t)d*2 + h, ex);
    }
}

// out[di] += xp_src[si] * softmax_weight   (one wave per edge, 256-wide message)
__global__ __launch_bounds__(256)
void scatter_msg(const int* __restrict__ si, const int* __restrict__ di,
                 const float* __restrict__ alpha, const float* __restrict__ sum,
                 const float* __restrict__ xp_src, float* __restrict__ out, int E)
{
    const int e    = blockIdx.x * (blockDim.x >> 5) + (threadIdx.x >> 5);
    const int lane = threadIdx.x & 31;
    if (e >= E) return;
    const int s = si[e], d = di[e];
    const float w0 = alpha[(size_t)e*2]     / (sum[(size_t)d*2]     + 1e-16f);
    const float w1 = alpha[(size_t)e*2 + 1] / (sum[(size_t)d*2 + 1] + 1e-16f);
    const float* xr  = xp_src + (size_t)s * 256;
    float* orow      = out    + (size_t)d * 256;
    #pragma unroll
    for (int it = 0; it < 8; ++it) {
        const int i = it * 32 + lane;
        const float w = (i < 128) ? w0 : w1;
        atomicAdd(orow + i, xr[i] * w);
    }
}

// semantic softmax over metapaths: beta = softmax_m( wacc[m] . q )
__global__ void beta_kernel(const float* __restrict__ wacc, const float* __restrict__ q,
                            float* __restrict__ beta, int M)
{
    const int lane = threadIdx.x;
    float sc[4];
    for (int m = 0; m < M; ++m) {
        float s = 0.f;
        for (int f = lane; f < 256; f += 32) s += wacc[m*256 + f] * q[f];
        #pragma unroll
        for (int off = 16; off; off >>= 1) s += __shfl_xor(s, off, 32);
        sc[m] = s;
    }
    if (lane == 0) {
        float mx = sc[0];
        for (int m = 1; m < M; ++m) mx = fmaxf(mx, sc[m]);
        float tot = 0.f;
        for (int m = 0; m < M; ++m) { sc[m] = __expf(sc[m] - mx); tot += sc[m]; }
        for (int m = 0; m < M; ++m) beta[m] = sc[m] / tot;
    }
}

// res = beta0*relu(out0) + beta1*relu(out1), written in place over out0
__global__ void combine2(float* __restrict__ out0, const float* __restrict__ out1,
                         const float* __restrict__ beta, int total)
{
    int i = blockIdx.x * blockDim.x + threadIdx.x;
    if (i >= total) return;
    const float r0 = fmaxf(out0[i], 0.f);
    const float r1 = fmaxf(out1[i], 0.f);
    out0[i] = beta[0] * r0 + beta[1] * r1;
}
// single metapath: softmax over 1 element == 1 -> res = relu(out0)
__global__ void combine1(float* __restrict__ out0, int total)
{
    int i = blockIdx.x * blockDim.x + threadIdx.x;
    if (i < total) out0[i] = fmaxf(out0[i], 0.f);
}

// logits = h[100k,256] @ w[256,20] + b ; one wave per node
__global__ __launch_bounds__(256)
void final_linear(const float* __restrict__ h, const float* __restrict__ w,
                  const float* __restrict__ bias, float* __restrict__ out, int N)
{
    const int node = blockIdx.x * (blockDim.x >> 5) + (threadIdx.x >> 5);
    const int lane = threadIdx.x & 31;
    if (node >= N) return;
    float acc[20];
    #pragma unroll
    for (int j = 0; j < 20; ++j) acc[j] = 0.f;
    for (int f = lane; f < 256; f += 32) {
        const float hv = h[(size_t)node * 256 + f];
        const float* wr = w + (size_t)f * 20;
        #pragma unroll
        for (int j = 0; j < 20; ++j) acc[j] += hv * wr[j];
    }
    #pragma unroll
    for (int j = 0; j < 20; ++j) {
        float v = acc[j];
        #pragma unroll
        for (int off = 16; off; off >>= 1) v += __shfl_xor(v, off, 32);
        if (lane == 0) out[(size_t)node * 20 + j] = v + bias[j];
    }
}

__global__ void logsoftmax_k(const float* __restrict__ logits, float* __restrict__ out, int N)
{
    int n = blockIdx.x * blockDim.x + threadIdx.x;
    if (n >= N) return;
    const float* r = logits + (size_t)n * 20;
    float mx = r[0];
    #pragma unroll
    for (int j = 1; j < 20; ++j) mx = fmaxf(mx, r[j]);
    float s = 0.f;
    #pragma unroll
    for (int j = 0; j < 20; ++j) s += __expf(r[j] - mx);
    const float lse = mx + logf(s);
    #pragma unroll
    for (int j = 0; j < 20; ++j) out[(size_t)n * 20 + j] = r[j] - lse;
}

// ---------------------------------------------------------------------------
// Host orchestration
// ---------------------------------------------------------------------------
extern "C" void kernel_launch(void* const* d_in, const int* in_sizes, int n_in,
                              void* d_out, int out_size, void* d_ws, size_t ws_size,
                              hipStream_t stream)
{
    const int NH = 100000, ND = 50000;
    const int E_HD = 500000, E_DH = 500000, E_HH = 600000;

    const float* x_hash = (const float*)d_in[0];
    const float* x_dom  = (const float*)d_in[1];
    const int*   ei_hd  = (const int*)d_in[2];
    const int*   ei_dh  = (const int*)d_in[3];
    const int*   ei_hh  = (const int*)d_in[4];
    // params flattened in dict insertion order:
    //   conv{1,2}: proj_w{H,D}, proj_b{H,D}, lin_src{hd,dh,hh}, lin_dst{hd,dh,hh}, k_w, k_b, q
    const int C1 = 5, C2 = 18;
    const float* lin_w = (const float*)d_in[31];
    const float* lin_b = (const float*)d_in[32];

    float* ws = (float*)d_ws;
    size_t off = 0;
    float* XP_H   = ws + off; off += (size_t)NH * 256;
    float* XP_D   = ws + off; off += (size_t)ND * 256;
    float* OUT_DH = ws + off; off += (size_t)NH * 256;  // metapath dh (dst Hash) / h_hash
    float* OUT_HH = ws + off; off += (size_t)NH * 256;  // metapath hh (dst Hash)
    float* OUT_HD = ws + off; off += (size_t)ND * 256;  // metapath hd (dst Domain) / h_dom
    float* ALPHA  = ws + off; off += (size_t)E_HH * 2;
    float* ASRC   = ws + off; off += (size_t)NH * 2;
    float* ADST   = ws + off; off += (size_t)NH * 2;
    unsigned* SEGMAX = (unsigned*)(ws + off); off += (size_t)NH * 2;
    float* SEGSUM = ws + off; off += (size_t)NH * 2;
    float* WACC   = ws + off; off += 512;
    float* BETA   = ws + off; off += 2;
    // bf16 fragment-packed weights (ushort), allocated in float units
    unsigned short* WP_C1H = (unsigned short*)(ws + off); off += (size_t)128 * 256 / 2;
    unsigned short* WP_C1D = (unsigned short*)(ws + off); off += (size_t)64  * 256 / 2;
    unsigned short* WP_C2H = (unsigned short*)(ws + off); off += (size_t)256 * 256 / 2;
    unsigned short* WP_C2D = (unsigned short*)(ws + off); off += (size_t)256 * 256 / 2;
    unsigned short* WP_KW1 = (unsigned short*)(ws + off); off += (size_t)256 * 256 / 2;
    unsigned short* WP_KW2 = (unsigned short*)(ws + off); off += (size_t)256 * 256 / 2;

    float nbig = -3.0e38f; unsigned nu; __builtin_memcpy(&nu, &nbig, 4);
    const unsigned negkey = ~nu;   // fkey(-3e38): lower than any attention logit

    auto pack = [&](int pidx, unsigned short* Wp, int K) {
        const int n = K * 256;
        pack_w_bf16<<<dim3((n + 255) / 256), dim3(256), 0, stream>>>(
            (const float*)d_in[pidx], Wp, n);
    };
    // pre-pack all six 256-col weights into WMMA B-fragment bf16 layout
    pack(C1 + 0,  WP_C1H, 128);
    pack(C1 + 1,  WP_C1D, 64);
    pack(C1 + 10, WP_KW1, 256);
    pack(C2 + 0,  WP_C2H, 256);
    pack(C2 + 1,  WP_C2D, 256);
    pack(C2 + 10, WP_KW2, 256);

    auto gemm = [&](const float* A, int M, int K, const unsigned short* Wp, const float* b,
                    float* Cout, float* wred, int reluA, int mode, float scale) {
        wmma_gemm_256<<<dim3((M + 15) / 16), dim3(128), 0, stream>>>(
            A, Wp, b, Cout, wred, M, K, reluA, mode, scale);
    };

    auto edge_phase = [&](const int* ei, int E,
                          const float* xp_src, int Nsrc,
                          const float* xp_dst, int Ndst,
                          const float* lsrc, const float* ldst, float* outbuf) {
        const int* si = ei;
        const int* di = ei + E;
        node_alpha<<<dim3((Nsrc + 7) / 8), dim3(256), 0, stream>>>(xp_src, lsrc, ASRC, Nsrc);
        node_alpha<<<dim3((Ndst + 7) / 8), dim3(256), 0, stream>>>(xp_dst, ldst, ADST, Ndst);
        fill_u32<<<dim3((Ndst * 2 + 255) / 256), dim3(256), 0, stream>>>(SEGMAX, negkey, Ndst * 2);
        fill_f32<<<dim3((Ndst * 2 + 255) / 256), dim3(256), 0, stream>>>(SEGSUM, 0.f, Ndst * 2);
        edge_max_k<<<dim3((E + 255) / 256), dim3(256), 0, stream>>>(si, di, ASRC, ADST, SEGMAX, E);
        edge_expsum_k<<<dim3((E + 255) / 256), dim3(256), 0, stream>>>(si, di, ASRC, ADST, SEGMAX, SEGSUM, ALPHA, E);
        const int tot = Ndst * 256;
        fill_f32<<<dim3((tot + 255) / 256), dim3(256), 0, stream>>>(outbuf, 0.f, tot);
        scatter_msg<<<dim3((E + 7) / 8), dim3(256), 0, stream>>>(si, di, ALPHA, SEGSUM, xp_src, outbuf, E);
    };

    auto conv = [&](const float* in_h, int Kh, const unsigned short* WpH,
                    const float* in_d, int Kd, const unsigned short* WpD,
                    const unsigned short* WpKW, int P) {
        const float* PBH   = (const float*)d_in[P + 2];
        const float* PBD   = (const float*)d_in[P + 3];
        const float* LS_HD = (const float*)d_in[P + 4];
        const float* LS_DH = (const float*)d_in[P + 5];
        const float* LS_HH = (const float*)d_in[P + 6];
        const float* LD_HD = (const float*)d_in[P + 7];
        const float* LD_DH = (const float*)d_in[P + 8];
        const float* LD_HH = (const float*)d_in[P + 9];
        const float* KB    = (const float*)d_in[P + 11];
        const float* Q     = (const float*)d_in[P + 12];

        // node-type projections (WMMA bf16)
        gemm(in_h, NH, Kh, WpH, PBH, XP_H, nullptr, 0, 0, 0.f);
        gemm(in_d, ND, Kd, WpD, PBD, XP_D, nullptr, 0, 0, 0.f);

        // per-edge-type attention + aggregation
        edge_phase(ei_hd, E_HD, XP_H, NH, XP_D, ND, LS_HD, LD_HD, OUT_HD);  // Hash->Domain
        edge_phase(ei_dh, E_DH, XP_D, ND, XP_H, NH, LS_DH, LD_DH, OUT_DH);  // Domain->Hash
        edge_phase(ei_hh, E_HH, XP_H, NH, XP_H, NH, LS_HH, LD_HH, OUT_HH);  // Hash->Hash

        // semantic attention, Hash (metapaths [dh, hh]); tanh-mean fused in GEMM
        fill_f32<<<dim3(2), dim3(256), 0, stream>>>(WACC, 0.f, 512);
        gemm(OUT_DH, NH, 256, WpKW, KB, nullptr, WACC,       1, 1, 1.0f / NH);
        gemm(OUT_HH, NH, 256, WpKW, KB, nullptr, WACC + 256, 1, 1, 1.0f / NH);
        beta_kernel<<<dim3(1), dim3(32), 0, stream>>>(WACC, Q, BETA, 2);
        combine2<<<dim3((NH * 256 + 255) / 256), dim3(256), 0, stream>>>(OUT_DH, OUT_HH, BETA, NH * 256);
        // Domain has one metapath -> beta == 1, just relu in place
        combine1<<<dim3((ND * 256 + 255) / 256), dim3(256), 0, stream>>>(OUT_HD, ND * 256);
    };

    conv(x_hash, 128, WP_C1H, x_dom, 64, WP_C1D, WP_KW1, C1);   // -> OUT_DH / OUT_HD
    conv(OUT_DH, 256, WP_C2H, OUT_HD, 256, WP_C2D, WP_KW2, C2); // second HAN layer

    float* logits = (float*)d_out;
    final_linear<<<dim3((NH + 7) / 8), dim3(256), 0, stream>>>(OUT_DH, lin_w, lin_b, logits, NH);
    logsoftmax_k<<<dim3((NH + 255) / 256), dim3(256), 0, stream>>>(logits, logits + (size_t)NH * 20, NH);
}
